// Block_48610439856264
// MI455X (gfx1250) — matrix operand order repaired and checked by reading, not compiled, not run on hardware
//
#include <hip/hip_runtime.h>
#include <math.h>

// ---------------------------------------------------------------------------
// CDNA5 (gfx1250) transformer block. All GEMMs + attention use
// v_wmma_f32_16x16x32_bf16 (wave32, f32 accumulate). Intermediates are bf16
// to halve HBM traffic (bandwidth-dominated workload at 23.3 TB/s).
// Global->LDS staging is register double-buffered so vmem loads overlap the
// WMMA stream; weight stream is prefetched ahead with global_prefetch_b8.
// Workspace requirement: ~128 MiB of d_ws.
// ---------------------------------------------------------------------------

typedef __attribute__((ext_vector_type(16))) __bf16 v16bf;
typedef __attribute__((ext_vector_type(8)))  float  v8f;
typedef unsigned short u16b;

union BF16x16 {
  v16bf v;
  u16b  u[16];
  uint4 q[2];
};
union BF16x8 {
  uint4 q;
  u16b  u[8];
};

__device__ __forceinline__ u16b f2b(float f) {               // f32 -> bf16 (RNE)
  union { float f; unsigned int i; } a; a.f = f;
  unsigned int r = a.i + 0x7FFFu + ((a.i >> 16) & 1u);
  return (u16b)(r >> 16);
}
__device__ __forceinline__ float b2f(u16b b) {               // bf16 -> f32
  union { unsigned int i; float f; } a; a.i = ((unsigned int)b) << 16; return a.f;
}

// ---------------------------------------------------------------------------
// LayerNorm: one wave per 512-wide row; 8 rows per 256-thread block.
// ---------------------------------------------------------------------------
__global__ __launch_bounds__(256) void ln_kernel(
    const float* __restrict__ x, const float* __restrict__ w,
    const float* __restrict__ b, u16b* __restrict__ out)
{
  const int wave = threadIdx.x >> 5, lane = threadIdx.x & 31;
  const int row  = blockIdx.x * 8 + wave;
  const float* px = x + (size_t)row * 512 + lane * 16;
  float v[16];
#pragma unroll
  for (int j = 0; j < 4; ++j) {
    float4 f = ((const float4*)px)[j];
    v[4*j+0] = f.x; v[4*j+1] = f.y; v[4*j+2] = f.z; v[4*j+3] = f.w;
  }
  float s = 0.0f;
#pragma unroll
  for (int i = 0; i < 16; ++i) s += v[i];
#pragma unroll
  for (int m = 16; m >= 1; m >>= 1) s += __shfl_xor(s, m, 32);
  const float mu = s * (1.0f / 512.0f);
  float q = 0.0f;
#pragma unroll
  for (int i = 0; i < 16; ++i) { float d = v[i] - mu; q += d * d; }
#pragma unroll
  for (int m = 16; m >= 1; m >>= 1) q += __shfl_xor(q, m, 32);
  const float rs = rsqrtf(q * (1.0f / 512.0f) + 1e-5f);

  BF16x16 t;
  const float* pw = w + lane * 16;
  const float* pb = b + lane * 16;
#pragma unroll
  for (int i = 0; i < 16; ++i)
    t.u[i] = f2b((v[i] - mu) * rs * pw[i] + pb[i]);
  uint4* dst = (uint4*)(out + (size_t)row * 512 + lane * 16);
  dst[0] = t.q[0]; dst[1] = t.q[1];
}

// ---------------------------------------------------------------------------
// Generic WMMA GEMM:  out = epi( A(bf16)[M,K] @ W(f32->bf16)[K,N] + bias )
// Block tile 128x128, K-step 32. 8 waves, each owns a 32(M)x64(N) sub-tile
// = 2x4 WMMA tiles of 16x16. W is converted & TRANSPOSED into LDS so the
// B-fragment (lane = N column, lane-half h = K octets 8h, 16+8h per ISA layout) is two
// contiguous 16B reads. Global tile k+1 is fetched into registers while the
// WMMAs for tile k run (register double-buffer); W stream prefetched ahead.
//   EPI 0: store bf16 (no bias)   EPI 1: +bias, exact-erf GELU, store bf16
//   EPI 2: +bias +resid(f32), store f32
// ---------------------------------------------------------------------------
template <int EPI>
__global__ __launch_bounds__(256) void gemm_wmma(
    const u16b*  __restrict__ A,    const float* __restrict__ W,
    const float* __restrict__ bias, const float* __restrict__ resid,
    float* __restrict__ outF, u16b* __restrict__ outB,
    int K, int N)
{
  __shared__ u16b As [128][40];   // 128 rows x 32 K (pad 8)
  __shared__ u16b WsT[128][40];   // transposed: [n][k] (pad 8)

  const int tid  = threadIdx.x;
  const int wave = tid >> 5, lane = tid & 31;
  const int lh   = lane >> 4, ln = lane & 15;
  const int bm   = blockIdx.y * 128, bn = blockIdx.x * 128;
  const int wm   = (wave >> 1) * 32, wn = (wave & 1) * 64;

  // staging assignments
  const int ar = tid >> 1, ac = (tid & 1) * 16;        // A: 2 thr/row, 16 bf16
  const int wr = tid >> 3, wc = (tid & 7) * 16;        // W: 8 thr/k-row, 16 f32
  const u16b*  aSrc = A + (size_t)(bm + ar) * K + ac;  // advance by +32 per k0
  const float* wSrc = W + (size_t)wr * N + bn + wc;    // advance by +32*N

  const v8f zero = {};
  v8f acc[2][4];
#pragma unroll
  for (int ms = 0; ms < 2; ++ms)
#pragma unroll
    for (int ns = 0; ns < 4; ++ns) acc[ms][ns] = zero;

  // prologue: fetch tile 0 into registers
  uint4  rA0 = *(const uint4*)aSrc;
  uint4  rA1 = *(const uint4*)(aSrc + 8);
  float4 rW[4];
#pragma unroll
  for (int j = 0; j < 4; ++j) rW[j] = ((const float4*)wSrc)[j];

  for (int k0 = 0; k0 < K; k0 += 32) {
    // stage registers -> LDS
    *(uint4*)&As[ar][ac]     = rA0;
    *(uint4*)&As[ar][ac + 8] = rA1;
#pragma unroll
    for (int j = 0; j < 4; ++j) {
      WsT[wc + j*4 + 0][wr] = f2b(rW[j].x);
      WsT[wc + j*4 + 1][wr] = f2b(rW[j].y);
      WsT[wc + j*4 + 2][wr] = f2b(rW[j].z);
      WsT[wc + j*4 + 3][wr] = f2b(rW[j].w);
    }
    __syncthreads();

    // fetch tile k0+32 into registers (overlaps the WMMAs below)
    if (k0 + 32 < K) {
      const u16b*  aN = aSrc + (k0 + 32);
      const float* wN = wSrc + (size_t)(k0 + 32) * N;
      rA0 = *(const uint4*)aN;
      rA1 = *(const uint4*)(aN + 8);
#pragma unroll
      for (int j = 0; j < 4; ++j) rW[j] = ((const float4*)wN)[j];
      if (k0 + 64 < K)                      // global_prefetch_b8, 2 tiles ahead
        __builtin_prefetch(wSrc + (size_t)(k0 + 64) * N, 0, 3);
    }

    v16bf af[2];
#pragma unroll
    for (int ms = 0; ms < 2; ++ms) {     // A frag: lane row = ln, K octets by lh
      BF16x16 t;
      const u16b* rowp = &As[wm + ms*16 + ln][0];
      t.q[0] = *(const uint4*)(rowp + lh*8);
      t.q[1] = *(const uint4*)(rowp + 16 + lh*8);
      af[ms] = t.v;
    }
    v16bf bfr[4];
#pragma unroll
    for (int ns = 0; ns < 4; ++ns) {     // B frag: lane col = ln, K octets by lh
      BF16x16 t;
      const u16b* colp = &WsT[wn + ns*16 + ln][0];
      t.q[0] = *(const uint4*)(colp + lh*8);
      t.q[1] = *(const uint4*)(colp + 16 + lh*8);
      bfr[ns] = t.v;
    }
#pragma unroll
    for (int ms = 0; ms < 2; ++ms)
#pragma unroll
      for (int ns = 0; ns < 4; ++ns)
        acc[ms][ns] = __builtin_amdgcn_wmma_f32_16x16x32_bf16(
            false, af[ms], false, bfr[ns], (short)0, acc[ms][ns], false, false);

    __syncthreads();                     // LDS reads done before next stage
  }

  // epilogue; C layout: VGPR r -> row = r + 8*lh, col = ln
#pragma unroll
  for (int ms = 0; ms < 2; ++ms) {
#pragma unroll
    for (int ns = 0; ns < 4; ++ns) {
      const int gr0 = bm + wm + ms*16 + lh*8;
      const int gc  = bn + wn + ns*16 + ln;
      float bc = 0.0f;
      if constexpr (EPI != 0) bc = bias[gc];
#pragma unroll
      for (int r = 0; r < 8; ++r) {
        size_t idx = (size_t)(gr0 + r) * N + gc;
        float v = acc[ms][ns][r] + bc;
        if constexpr (EPI == 1) {
          v = 0.5f * v * (1.0f + erff(v * 0.70710678118654752f));
          outB[idx] = f2b(v);
        } else if constexpr (EPI == 2) {
          outF[idx] = v + resid[idx];
        } else {
          outB[idx] = f2b(v);
        }
      }
    }
  }
}

// ---------------------------------------------------------------------------
// ks = k + s (bf16), k lives at qkv[:, 512:1024]. 8 elements per thread.
// ---------------------------------------------------------------------------
__global__ __launch_bounds__(256) void addks_kernel(
    const u16b* __restrict__ qkv, const u16b* __restrict__ s, u16b* __restrict__ ks)
{
  int i = (blockIdx.x * 256 + threadIdx.x) * 8;   // 8192*512 elements exactly
  int row = i >> 9, col = i & 511;
  BF16x8 a, b, o;
  a.q = *(const uint4*)&qkv[(size_t)row * 1536 + 512 + col];
  b.q = *(const uint4*)&s[i];
#pragma unroll
  for (int j = 0; j < 8; ++j) o.u[j] = f2b(b2f(a.u[j]) + b2f(b.u[j]));
  *(uint4*)&ks[i] = o.q;
}

// ---------------------------------------------------------------------------
// Flash attention per (b,h). Block = 128 Q rows (16 per wave).
// Per 32-key tile: cooperative LDS stage of Ks (key-major) and V (transposed),
// QK^T (4 WMMA), online softmax (half-wave shfl reductions), P->LDS->A-frag,
// P@V (4 WMMA). K/V tile kt+1 fetched into registers during compute.
// Epilogue: O/l * gate -> og (bf16).
// ---------------------------------------------------------------------------
__global__ __launch_bounds__(256) void attn_kernel(
    const u16b* __restrict__ qkv, const u16b* __restrict__ ks,
    const u16b* __restrict__ gate, u16b* __restrict__ og)
{
  __shared__ u16b Kls[32][72];      // [key][d]   (pad to 72)
  __shared__ u16b VT [64][40];      // [d][key]   (pad to 40)
  __shared__ u16b Pls[8][16][40];   // per-wave P patch

  const int tid  = threadIdx.x;
  const int wave = tid >> 5, lane = tid & 31;
  const int lh   = lane >> 4, ln = lane & 15;
  const int bh   = blockIdx.y;
  const int bb   = bh >> 3, h = bh & 7;
  const int tokBase = bb * 2048;
  const int qRow = blockIdx.x * 128 + wave * 16;     // within this batch

  // staging assignment: 8 threads per key row, 8 bf16 each
  const int skey = tid >> 3, sc8 = (tid & 7) * 8;
  const u16b* kBase = ks  + (size_t)(tokBase + skey) * 512  + h * 64 + sc8;
  const u16b* vBase = qkv + (size_t)(tokBase + skey) * 1536 + 1024 + h * 64 + sc8;

  // Q A-fragments, d = 0..63 (two K=32 fragments)
  v16bf qf[2];
  {
    const u16b* qp = qkv + (size_t)(tokBase + qRow + ln) * 1536 + h * 64;
#pragma unroll
    for (int f = 0; f < 2; ++f) {
      BF16x16 t;
      int k0 = f * 32 + lh * 8;
      t.q[0] = *(const uint4*)(qp + k0);
      t.q[1] = *(const uint4*)(qp + k0 + 16);
      qf[f] = t.v;
    }
  }

  const v8f zero = {};
  v8f o[4] = {zero, zero, zero, zero};
  float m_i[8], l_i[8];
#pragma unroll
  for (int r = 0; r < 8; ++r) { m_i[r] = -3.0e38f; l_i[r] = 0.0f; }

  // prologue: fetch key tile 0 into registers
  uint4 rK = *(const uint4*)kBase;
  uint4 rV = *(const uint4*)vBase;

  for (int kt = 0; kt < 2048; kt += 32) {
    // stage registers -> LDS
    *(uint4*)&Kls[skey][sc8] = rK;
    {
      BF16x8 t; t.q = rV;
#pragma unroll
      for (int j = 0; j < 8; ++j) VT[sc8 + j][skey] = t.u[j];
    }
    __syncthreads();

    // fetch key tile kt+32 (overlaps compute below)
    if (kt + 32 < 2048) {
      rK = *(const uint4*)(kBase + (size_t)(kt + 32) * 512);
      rV = *(const uint4*)(vBase + (size_t)(kt + 32) * 1536);
    }

    // scores: S[16x32] = Q @ Ks^T * 0.125  (B frag: lane col = key, K = d)
    v8f sc[2];
#pragma unroll
    for (int ns = 0; ns < 2; ++ns) {
      BF16x16 t0, t1;
      const u16b* kr = &Kls[ns*16 + ln][0];
      t0.q[0] = *(const uint4*)(kr + lh*8);
      t0.q[1] = *(const uint4*)(kr + 16 + lh*8);
      t1.q[0] = *(const uint4*)(kr + 32 + lh*8);
      t1.q[1] = *(const uint4*)(kr + 32 + 16 + lh*8);
      v8f c = zero;
      c = __builtin_amdgcn_wmma_f32_16x16x32_bf16(false, qf[0], false, t0.v, (short)0, c, false, false);
      c = __builtin_amdgcn_wmma_f32_16x16x32_bf16(false, qf[1], false, t1.v, (short)0, c, false, false);
      sc[ns] = c * 0.125f;
    }

    // online softmax (row = r + 8*lh lives across the 16-lane half)
    float p0[8], p1[8], alpha[8];
#pragma unroll
    for (int r = 0; r < 8; ++r) {
      float rm = fmaxf(sc[0][r], sc[1][r]);
#pragma unroll
      for (int m = 8; m >= 1; m >>= 1) rm = fmaxf(rm, __shfl_xor(rm, m, 32));
      float mn = fmaxf(m_i[r], rm);
      alpha[r] = __expf(m_i[r] - mn);
      p0[r] = __expf(sc[0][r] - mn);
      p1[r] = __expf(sc[1][r] - mn);
      float rsum = p0[r] + p1[r];
#pragma unroll
      for (int m = 8; m >= 1; m >>= 1) rsum += __shfl_xor(rsum, m, 32);
      l_i[r] = l_i[r] * alpha[r] + rsum;
      m_i[r] = mn;
    }
#pragma unroll
    for (int nt = 0; nt < 4; ++nt)
#pragma unroll
      for (int r = 0; r < 8; ++r) o[nt][r] *= alpha[r];

    // P (C-layout) -> LDS -> A-fragment. Same-wave DS ops are in-order on
    // CDNA5; the asm barrier stops compiler reordering of the LDS write/read.
#pragma unroll
    for (int r = 0; r < 8; ++r) {
      Pls[wave][r + 8*lh][ln]      = f2b(p0[r]);
      Pls[wave][r + 8*lh][16 + ln] = f2b(p1[r]);
    }
    asm volatile("" ::: "memory");
    BF16x16 pa;
    {
      const u16b* pr = &Pls[wave][ln][0];
      pa.q[0] = *(const uint4*)(pr + lh*8);
      pa.q[1] = *(const uint4*)(pr + 16 + lh*8);
    }
    // O += P @ V  (B frag from transposed V: contiguous along key)
#pragma unroll
    for (int nt = 0; nt < 4; ++nt) {
      BF16x16 vb;
      const u16b* vr = &VT[nt*16 + ln][0];
      vb.q[0] = *(const uint4*)(vr + lh*8);
      vb.q[1] = *(const uint4*)(vr + 16 + lh*8);
      o[nt] = __builtin_amdgcn_wmma_f32_16x16x32_bf16(
          false, pa.v, false, vb.v, (short)0, o[nt], false, false);
    }
    __syncthreads();                 // Kls/VT reads done before next stage
  }

  // epilogue: O/l * gate -> og (bf16)
#pragma unroll
  for (int nt = 0; nt < 4; ++nt) {
#pragma unroll
    for (int r = 0; r < 8; ++r) {
      int tokR = tokBase + qRow + r + 8*lh;
      int gc   = h * 64 + nt*16 + ln;
      size_t idx = (size_t)tokR * 512 + gc;
      og[idx] = f2b((o[nt][r] / l_i[r]) * b2f(gate[idx]));
    }
  }
}

// ---------------------------------------------------------------------------
extern "C" void kernel_launch(void* const* d_in, const int* in_sizes, int n_in,
                              void* d_out, int out_size, void* d_ws, size_t ws_size,
                              hipStream_t stream) {
  (void)in_sizes; (void)n_in; (void)out_size; (void)ws_size;
  const float* x      = (const float*)d_in[0];
  const float* e      = (const float*)d_in[1];
  const float* w_qkv  = (const float*)d_in[2];
  const float* w_s    = (const float*)d_in[3];
  const float* w_gate = (const float*)d_in[4];
  const float* w_proj = (const float*)d_in[5];
  const float* b_proj = (const float*)d_in[6];
  const float* ln1_w  = (const float*)d_in[7];
  const float* ln1_b  = (const float*)d_in[8];
  const float* ln2_w  = (const float*)d_in[9];
  const float* ln2_b  = (const float*)d_in[10];
  const float* ln3_w  = (const float*)d_in[11];
  const float* ln3_b  = (const float*)d_in[12];
  const float* w_fc1  = (const float*)d_in[13];
  const float* b_fc1  = (const float*)d_in[14];
  const float* w_fc2  = (const float*)d_in[15];
  const float* b_fc2  = (const float*)d_in[16];

  const size_t T = 8192;            // B*N tokens
  char* p = (char*)d_ws;
  u16b*  xn    = (u16b*)p;  p += T * 512 * 2;
  u16b*  en    = (u16b*)p;  p += T * 512 * 2;
  u16b*  qkvb  = (u16b*)p;  p += T * 1536 * 2;
  u16b*  sb    = (u16b*)p;  p += T * 512 * 2;
  u16b*  gateb = (u16b*)p;  p += T * 512 * 2;
  u16b*  ksb   = (u16b*)p;  p += T * 512 * 2;
  u16b*  ogb   = (u16b*)p;  p += T * 512 * 2;
  float* x1    = (float*)p; p += T * 512 * 4;
  u16b*  xn3   = (u16b*)p;  p += T * 512 * 2;
  u16b*  hbuf  = (u16b*)p;  p += T * 2048 * 2;

  // 1) LayerNorms of x and e -> bf16
  ln_kernel<<<1024, 256, 0, stream>>>(x, ln1_w, ln1_b, xn);
  ln_kernel<<<1024, 256, 0, stream>>>(e, ln2_w, ln2_b, en);

  // 2) qkv / s / gate GEMMs (bf16 outputs)
  gemm_wmma<0><<<dim3(12, 64), 256, 0, stream>>>(xn, w_qkv, nullptr, nullptr,
                                                 nullptr, qkvb, 512, 1536);
  gemm_wmma<0><<<dim3(4, 64), 256, 0, stream>>>(en, w_s, nullptr, nullptr,
                                                nullptr, sb, 512, 512);
  gemm_wmma<0><<<dim3(4, 64), 256, 0, stream>>>(xn, w_gate, nullptr, nullptr,
                                                nullptr, gateb, 512, 512);

  // 3) ks = k + s
  addks_kernel<<<2048, 256, 0, stream>>>(qkvb, sb, ksb);

  // 4) flash attention + gate -> og
  attn_kernel<<<dim3(16, 32), 256, 0, stream>>>(qkvb, ksb, gateb, ogb);

  // 5) x1 = x + og @ w_proj + b_proj   (f32)
  gemm_wmma<2><<<dim3(4, 64), 256, 0, stream>>>(ogb, w_proj, b_proj, x,
                                                x1, nullptr, 512, 512);

  // 6) MLP branch
  ln_kernel<<<1024, 256, 0, stream>>>(x1, ln3_w, ln3_b, xn3);
  gemm_wmma<1><<<dim3(16, 64), 256, 0, stream>>>(xn3, w_fc1, b_fc1, nullptr,
                                                 nullptr, hbuf, 512, 2048);
  gemm_wmma<2><<<dim3(4, 64), 256, 0, stream>>>(hbuf, w_fc2, b_fc2, x1,
                                                (float*)d_out, nullptr, 2048, 512);
}
